// SelfAttention_22050362097926
// MI455X (gfx1250) — compile-verified, matrix operand order checked by reading, not run on hardware
//
#include <hip/hip_runtime.h>
#include <hip/hip_bf16.h>

typedef __bf16 bf16;
typedef __attribute__((ext_vector_type(16))) __bf16 v16bf;
typedef __attribute__((ext_vector_type(8)))  float  v8f;
typedef __attribute__((ext_vector_type(4)))  int    v4i;

#define S_TXT  256
#define S_IMG  3072
#define S_TOT  3328
#define DM     1920
#define NH     30
#define HD     64
#define NQKV   5760   // 3*DM

#if defined(__has_builtin)
#if __has_builtin(__builtin_amdgcn_global_load_async_to_lds_b128)
#define HAVE_ASYNC_LDS 1
#endif
#endif

typedef __attribute__((address_space(1))) v4i gv4i;   // global 16B vector
typedef __attribute__((address_space(3))) v4i lv4i;   // LDS 16B vector

static __device__ __forceinline__ v16bf ld16(const bf16* p) {
    return *reinterpret_cast<const v16bf*>(p);
}

// 16-byte global -> LDS copy; async (ASYNCcnt) when the builtin exists,
// otherwise load+ds_store fallback (still de-duplicates across waves).
static __device__ __forceinline__ void cp16_g2l(const bf16* gp, bf16* lp) {
#if defined(HAVE_ASYNC_LDS)
    __builtin_amdgcn_global_load_async_to_lds_b128(
        (gv4i*)gp, (lv4i*)lp, /*offset=*/0, /*cpol=*/0);
#else
    *reinterpret_cast<v4i*>(lp) = *reinterpret_cast<const v4i*>(gp);
#endif
}

static __device__ __forceinline__ void wait_async_done() {
#if defined(HAVE_ASYNC_LDS)
#if __has_builtin(__builtin_amdgcn_s_wait_asynccnt)
    __builtin_amdgcn_s_wait_asynccnt(0);
#else
    asm volatile("s_wait_asynccnt 0x0" ::: "memory");
#endif
#endif
}

// ---------------------------------------------------------------------------
// Kernel 1: X = concat(enc, hid) -> bf16
// ---------------------------------------------------------------------------
__global__ void __launch_bounds__(256)
pack_x_kernel(const float* __restrict__ hid, const float* __restrict__ enc,
              bf16* __restrict__ Xb)
{
    size_t idx = (size_t)blockIdx.x * 256 + threadIdx.x;
    if (idx >= (size_t)S_TOT * DM) return;
    int s = (int)(idx / DM);
    int c = (int)(idx % DM);
    float v = (s < S_TXT) ? enc[(size_t)s * DM + c]
                          : hid[(size_t)(s - S_TXT) * DM + c];
    Xb[idx] = (bf16)v;
}

// ---------------------------------------------------------------------------
// Kernel 2: pack Wq|Wk|Wv -> bf16 [5760][1920], Wo -> bf16, concat biases
// ---------------------------------------------------------------------------
__global__ void __launch_bounds__(256)
pack_w_kernel(const float* __restrict__ Wq, const float* __restrict__ bq,
              const float* __restrict__ Wk, const float* __restrict__ bk,
              const float* __restrict__ Wv, const float* __restrict__ bv,
              const float* __restrict__ Wo,
              bf16* __restrict__ Wqkvb, bf16* __restrict__ Wob,
              float* __restrict__ bqkv)
{
    const size_t NW_QKV = (size_t)NQKV * DM;
    const size_t NW_O   = (size_t)DM * DM;
    size_t i = (size_t)blockIdx.x * 256 + threadIdx.x;
    if (i < NW_QKV) {
        int r = (int)(i / DM), c = (int)(i % DM);
        float v;
        if      (r < DM)     v = Wq[(size_t)r * DM + c];
        else if (r < 2 * DM) v = Wk[(size_t)(r - DM) * DM + c];
        else                 v = Wv[(size_t)(r - 2 * DM) * DM + c];
        Wqkvb[i] = (bf16)v;
    } else if (i < NW_QKV + NW_O) {
        size_t j = i - NW_QKV;
        Wob[j] = (bf16)Wo[j];
    } else if (i < NW_QKV + NW_O + NQKV) {
        int r = (int)(i - NW_QKV - NW_O);
        float v;
        if      (r < DM)     v = bq[r];
        else if (r < 2 * DM) v = bk[r - DM];
        else                 v = bv[r - 2 * DM];
        bqkv[r] = v;
    }
}

// ---------------------------------------------------------------------------
// Kernel 3: C[M][N] = A[M][K] @ W[N][K]^T + bias  (bf16 in, f32 out)
// 128 threads = 4 waves; block tile 128x128; wave tile 64x64 (4x4 WMMA accs).
// Optional epilogue row-remap: m<T -> m+S_IMG, m>=T -> m-T (fused out split).
// ---------------------------------------------------------------------------
__global__ void __launch_bounds__(128)
gemm_bf16_nt(const bf16* __restrict__ A, const bf16* __restrict__ W,
             const float* __restrict__ bias, float* __restrict__ C,
             int M, int N, int K, int remapT)
{
    const int lane = threadIdx.x & 31;
    const int wid  = threadIdx.x >> 5;
    const int wm = blockIdx.y * 128 + (wid >> 1) * 64;
    const int wn = blockIdx.x * 128 + (wid & 1) * 64;
    const int rsel = lane & 15;            // row within 16-tile
    const int ksel = (lane >> 4) * 16;     // which K-half this lane owns

    const bf16* Ap[4];
    const bf16* Wp[4];
#pragma unroll
    for (int i = 0; i < 4; ++i) {
        Ap[i] = A + (size_t)(wm + i * 16 + rsel) * K + ksel;
        Wp[i] = W + (size_t)(wn + i * 16 + rsel) * K + ksel;
    }

    const v8f zero8 = {0.f, 0.f, 0.f, 0.f, 0.f, 0.f, 0.f, 0.f};
    v8f acc[4][4];
#pragma unroll
    for (int i = 0; i < 4; ++i)
#pragma unroll
        for (int j = 0; j < 4; ++j) acc[i][j] = zero8;

    for (int k0 = 0; k0 < K; k0 += 32) {
        v16bf av[4], wv[4];
#pragma unroll
        for (int i = 0; i < 4; ++i) av[i] = ld16(Ap[i] + k0);
#pragma unroll
        for (int j = 0; j < 4; ++j) wv[j] = ld16(Wp[j] + k0);
#pragma unroll
        for (int i = 0; i < 4; ++i)
#pragma unroll
            for (int j = 0; j < 4; ++j)
                acc[i][j] = __builtin_amdgcn_wmma_f32_16x16x32_bf16(
                    false, av[i], false, wv[j], (short)0, acc[i][j], false, false);
    }

    const int col  = lane & 15;
    const int rofs = (lane >> 4) * 8;      // C layout: lanes>=16 hold rows +8
#pragma unroll
    for (int i = 0; i < 4; ++i)
#pragma unroll
        for (int j = 0; j < 4; ++j) {
            int n = wn + j * 16 + col;
            float b = bias[n];
#pragma unroll
            for (int r = 0; r < 8; ++r) {
                int m = wm + i * 16 + rofs + r;
                int om = m;
                if (remapT > 0) om = (m < remapT) ? (m + S_IMG) : (m - remapT);
                C[(size_t)om * N + n] = acc[i][j][r] + b;
            }
        }
}

// ---------------------------------------------------------------------------
// Kernel 4: per-(token,head) fp32 LayerNorm on q,k; RoPE on image rows;
// emit Q,K bf16 as [H][S][64] (frag-friendly) and V transposed [H][64][S].
// One wave per (s,h): lane owns d=lane and d=lane+32. wave32 xor-reductions.
// ---------------------------------------------------------------------------
__global__ void __launch_bounds__(256)
ln_rope_pack(const float* __restrict__ QKV,
             const float* __restrict__ cosb, const float* __restrict__ sinb,
             bf16* __restrict__ Qb, bf16* __restrict__ Kb, bf16* __restrict__ Vt)
{
    const int w    = blockIdx.x * 8 + (threadIdx.x >> 5);
    const int lane = threadIdx.x & 31;
    const int s = w / NH;
    const int h = w % NH;

    const float* row = QKV + (size_t)s * NQKV + h * HD;
    float q0 = row[lane],          q1 = row[lane + 32];
    float k0 = row[DM + lane],     k1 = row[DM + lane + 32];
    float v0 = row[2 * DM + lane], v1 = row[2 * DM + lane + 32];

    auto lnorm = [&](float& a, float& b) {
        float sum = a + b;
#pragma unroll
        for (int m = 1; m < 32; m <<= 1) sum += __shfl_xor(sum, m, 32);
        float mu = sum * (1.f / 64.f);
        float d0 = a - mu, d1 = b - mu;
        float vs = d0 * d0 + d1 * d1;
#pragma unroll
        for (int m = 1; m < 32; m <<= 1) vs += __shfl_xor(vs, m, 32);
        float inv = rsqrtf(vs * (1.f / 64.f) + 1e-5f);
        a = d0 * inv; b = d1 * inv;
    };
    lnorm(q0, q1);
    lnorm(k0, k1);

    if (s >= S_TXT) {  // RoPE on image tokens only
        int si = s - S_TXT;
        float c0 = cosb[(size_t)si * HD + lane];
        float c1 = cosb[(size_t)si * HD + lane + 32];
        float s0 = sinb[(size_t)si * HD + lane];
        float s1 = sinb[(size_t)si * HD + lane + 32];
        float nq0 = q0 * c0 - q1 * s0, nq1 = q1 * c1 + q0 * s1;
        float nk0 = k0 * c0 - k1 * s0, nk1 = k1 * c1 + k0 * s1;
        q0 = nq0; q1 = nq1; k0 = nk0; k1 = nk1;
    }

    size_t qo = ((size_t)h * S_TOT + s) * HD;
    Qb[qo + lane]      = (bf16)q0;
    Qb[qo + lane + 32] = (bf16)q1;
    Kb[qo + lane]      = (bf16)k0;
    Kb[qo + lane + 32] = (bf16)k1;
    Vt[((size_t)h * HD + lane)      * S_TOT + s] = (bf16)v0;
    Vt[((size_t)h * HD + lane + 32) * S_TOT + s] = (bf16)v1;
}

// ---------------------------------------------------------------------------
// Kernel 5: flash attention. grid=(S/256, H); 8 waves; wave owns 32 q-rows.
// K/V tiles (4 KB each) staged once per block into double-buffered LDS via
// async global->LDS copies (ASYNCcnt), overlapped with WMMA + softmax of the
// current tile. P round-trips through a per-wave LDS tile to convert the
// WMMA C-layout into A-frag layout for the PV matmul.
// ---------------------------------------------------------------------------
__global__ void __launch_bounds__(256)
flash_attn(const bf16* __restrict__ Qb, const bf16* __restrict__ Kb,
           const bf16* __restrict__ Vt, bf16* __restrict__ Ob)
{
    __shared__ bf16 ldsK[2][32 * 64];    // [buf][key][d]   4 KB each
    __shared__ bf16 ldsV[2][64 * 32];    // [buf][d][key]   4 KB each
    __shared__ bf16 ldsP[8 * 32 * 32];   // per-wave P tile 2 KB each

    const int tid  = threadIdx.x;
    const int lane = tid & 31;
    const int wid  = tid >> 5;
    const int h    = blockIdx.y;
    const int qm   = blockIdx.x * 256 + wid * 32;
    const int rsel = lane & 15;
    const int ksel = (lane >> 4) * 16;
    const int rofs = (lane >> 4) * 8;
    const float csc = 0.125f * 1.44269504088896340736f;  // 1/sqrt(64) * log2(e)

    const bf16* Qh = Qb + (size_t)h * S_TOT * HD;
    const bf16* Kh = Kb + (size_t)h * S_TOT * HD;
    const bf16* Vh = Vt + (size_t)h * HD * S_TOT;
    bf16* myP = ldsP + wid * 1024;

    // cooperative stage of one 32-key K tile + V tile into LDS buffer `buf`
    const int krow = tid >> 3, kseg = (tid & 7) * 8;   // 32 rows x 128B
    const int vrow = tid >> 2, vseg = (tid & 3) * 8;   // 64 rows x 64B
    auto stage = [&](int buf, int kb) {
        cp16_g2l(Kh + (size_t)(kb + krow) * HD + kseg, &ldsK[buf][krow * 64 + kseg]);
        cp16_g2l(Vh + (size_t)vrow * S_TOT + kb + vseg, &ldsV[buf][vrow * 32 + vseg]);
    };

    stage(0, 0);   // prefetch first tile before doing register setup

    v16bf qf[2][2];
#pragma unroll
    for (int t = 0; t < 2; ++t)
#pragma unroll
        for (int kk = 0; kk < 2; ++kk)
            qf[t][kk] = ld16(Qh + (size_t)(qm + t * 16 + rsel) * HD + kk * 32 + ksel);

    const v8f zero8 = {0.f, 0.f, 0.f, 0.f, 0.f, 0.f, 0.f, 0.f};
    v8f acco[2][4];
    float mi[2][8], li[2][8];
#pragma unroll
    for (int t = 0; t < 2; ++t) {
#pragma unroll
        for (int j = 0; j < 4; ++j) acco[t][j] = zero8;
#pragma unroll
        for (int r = 0; r < 8; ++r) { mi[t][r] = -3.0e38f; li[t][r] = 0.f; }
    }

    wait_async_done();
    __syncthreads();

    int buf = 0;
    for (int kb = 0; kb < S_TOT; kb += 32, buf ^= 1) {
        if (kb + 32 < S_TOT) stage(buf ^ 1, kb + 32);   // async next tile

        // ---- scores = Q @ K^T (K frags from LDS; scale folded into exp2) ----
        v16bf kf[2][2];
#pragma unroll
        for (int j2 = 0; j2 < 2; ++j2)
#pragma unroll
            for (int kk = 0; kk < 2; ++kk)
                kf[j2][kk] = *reinterpret_cast<const v16bf*>(
                    &ldsK[buf][(j2 * 16 + rsel) * 64 + kk * 32 + ksel]);

        v8f sc[2][2];
#pragma unroll
        for (int t = 0; t < 2; ++t)
#pragma unroll
            for (int j2 = 0; j2 < 2; ++j2) {
                v8f a = zero8;
                a = __builtin_amdgcn_wmma_f32_16x16x32_bf16(
                        false, qf[t][0], false, kf[j2][0], (short)0, a, false, false);
                a = __builtin_amdgcn_wmma_f32_16x16x32_bf16(
                        false, qf[t][1], false, kf[j2][1], (short)0, a, false, false);
                sc[t][j2] = a;
            }

        // ---- online softmax; P -> per-wave LDS tile (bf16) ----
#pragma unroll
        for (int t = 0; t < 2; ++t)
#pragma unroll
            for (int r = 0; r < 8; ++r) {
                float s0 = sc[t][0][r], s1 = sc[t][1][r];
                float mx = fmaxf(s0, s1);
#pragma unroll
                for (int m = 1; m < 16; m <<= 1) mx = fmaxf(mx, __shfl_xor(mx, m, 32));
                float mold = mi[t][r];
                float mnew = fmaxf(mold, mx);
                float alpha = exp2f((mold - mnew) * csc);
                float p0 = exp2f((s0 - mnew) * csc);
                float p1 = exp2f((s1 - mnew) * csc);
                float ps = p0 + p1;
#pragma unroll
                for (int m = 1; m < 16; m <<= 1) ps += __shfl_xor(ps, m, 32);
                li[t][r] = li[t][r] * alpha + ps;
                mi[t][r] = mnew;
#pragma unroll
                for (int j = 0; j < 4; ++j) acco[t][j][r] *= alpha;
                int prow = t * 16 + rofs + r;
                myP[prow * 32 + (lane & 15)]      = (bf16)p0;
                myP[prow * 32 + 16 + (lane & 15)] = (bf16)p1;
            }

        // ---- out += P @ V  (P as A-frags, V frags from LDS) ----
        v16bf pf[2];
#pragma unroll
        for (int t = 0; t < 2; ++t)
            pf[t] = *reinterpret_cast<const v16bf*>(myP + (t * 16 + rsel) * 32 + ksel);
        v16bf vf[4];
#pragma unroll
        for (int j = 0; j < 4; ++j)
            vf[j] = *reinterpret_cast<const v16bf*>(
                &ldsV[buf][(j * 16 + rsel) * 32 + ksel]);
#pragma unroll
        for (int t = 0; t < 2; ++t)
#pragma unroll
            for (int j = 0; j < 4; ++j)
                acco[t][j] = __builtin_amdgcn_wmma_f32_16x16x32_bf16(
                    false, pf[t], false, vf[j], (short)0, acco[t][j], false, false);

        wait_async_done();   // my async writes to buf^1 are done
        __syncthreads();     // everyone's writes done, everyone done reading buf
    }

    // ---- epilogue: normalize and emit attention output [S][DM] bf16 ----
#pragma unroll
    for (int t = 0; t < 2; ++t)
#pragma unroll
        for (int r = 0; r < 8; ++r) {
            float invl = 1.f / li[t][r];
            int srow = qm + t * 16 + rofs + r;
#pragma unroll
            for (int j = 0; j < 4; ++j)
                Ob[(size_t)srow * DM + h * HD + j * 16 + (lane & 15)] =
                    (bf16)(acco[t][j][r] * invl);
        }
}

// ---------------------------------------------------------------------------
extern "C" void kernel_launch(void* const* d_in, const int* in_sizes, int n_in,
                              void* d_out, int out_size, void* d_ws, size_t ws_size,
                              hipStream_t stream)
{
    const float* hid = (const float*)d_in[0];
    const float* enc = (const float*)d_in[1];
    const float* rc  = (const float*)d_in[2];
    const float* rs  = (const float*)d_in[3];
    const float* Wq  = (const float*)d_in[4];
    const float* bq  = (const float*)d_in[5];
    const float* Wk  = (const float*)d_in[6];
    const float* bk  = (const float*)d_in[7];
    const float* Wv  = (const float*)d_in[8];
    const float* bv  = (const float*)d_in[9];
    const float* Wo  = (const float*)d_in[10];
    const float* bo  = (const float*)d_in[11];
    float* out = (float*)d_out;

    char* ws = (char*)d_ws;
    size_t off = 0;
    auto alloc = [&](size_t bytes) { char* p = ws + off; off = (off + bytes + 255) & ~(size_t)255; return p; };
    bf16*  Xb    = (bf16*)alloc((size_t)S_TOT * DM * 2);
    bf16*  Wqkvb = (bf16*)alloc((size_t)NQKV * DM * 2);
    bf16*  Wob   = (bf16*)alloc((size_t)DM * DM * 2);
    float* bqkv  = (float*)alloc((size_t)NQKV * 4);
    float* QKVf  = (float*)alloc((size_t)S_TOT * NQKV * 4);
    bf16*  Qb    = (bf16*)alloc((size_t)NH * S_TOT * HD * 2);
    bf16*  Kb    = (bf16*)alloc((size_t)NH * S_TOT * HD * 2);
    bf16*  Vt    = (bf16*)alloc((size_t)NH * HD * S_TOT * 2);
    bf16*  Ob    = (bf16*)alloc((size_t)S_TOT * DM * 2);

    // 1) pack activations
    {
        size_t n = (size_t)S_TOT * DM;
        pack_x_kernel<<<dim3((unsigned)((n + 255) / 256)), dim3(256), 0, stream>>>(hid, enc, Xb);
    }
    // 2) pack weights + biases
    {
        size_t n = (size_t)(NQKV + DM) * DM + NQKV;
        pack_w_kernel<<<dim3((unsigned)((n + 255) / 256)), dim3(256), 0, stream>>>(
            Wq, bq, Wk, bk, Wv, bv, Wo, Wqkvb, Wob, bqkv);
    }
    // 3) fused QKV projection: [3328 x 1920] @ [5760 x 1920]^T -> f32
    gemm_bf16_nt<<<dim3(NQKV / 128, S_TOT / 128), dim3(128), 0, stream>>>(
        Xb, Wqkvb, bqkv, QKVf, S_TOT, NQKV, DM, 0);
    // 4) LayerNorm + RoPE + head-major repack (V transposed)
    ln_rope_pack<<<dim3((S_TOT * NH) / 8), dim3(256), 0, stream>>>(
        QKVf, rc, rs, Qb, Kb, Vt);
    // 5) flash attention (LDS-staged K/V, async double buffering)
    flash_attn<<<dim3(S_TOT / 256, NH), dim3(256), 0, stream>>>(Qb, Kb, Vt, Ob);
    // 6) output projection, (img, txt) split fused via row remap
    gemm_bf16_nt<<<dim3(DM / 128, S_TOT / 128), dim3(128), 0, stream>>>(
        Ob, Wob, bo, out, S_TOT, DM, DM, S_TXT);
}
